// SelfAttention_9388798509188
// MI455X (gfx1250) — compile-verified, matrix operand order checked by reading, not run on hardware
//
#include <hip/hip_runtime.h>
#include <hip/hip_bf16.h>
#include <math.h>

// Self-attention fwd for B=2, T=2048, D_MODEL=1024, HEADS=16, D_HEAD=64.
// bf16 WMMA (v_wmma_f32_16x16x32_bf16) everywhere, f32 accumulate.
// K/V tiles staged into LDS by the Tensor Data Mover (tensor_load_to_lds,
// TENSORcnt double-buffering) in the flash-attention kernel.
// Workspace (bf16): Q[32][2048][64] | K[32][2048][64] | Vt[32][64][2048] | Attn[4096][1024]
// => 4 * 8 MiB = 32 MiB required in d_ws.

typedef __bf16 bf16;
typedef __attribute__((ext_vector_type(16))) __bf16 v16bf;
typedef __attribute__((ext_vector_type(8)))  __bf16 v8bf;
typedef __attribute__((ext_vector_type(8)))  float   v8f;
typedef unsigned int v4u __attribute__((ext_vector_type(4)));
typedef int          v4i __attribute__((ext_vector_type(4)));
typedef int          v8i __attribute__((ext_vector_type(8)));

constexpr int Bc  = 2;
constexpr int Tc  = 2048;
constexpr int Hc  = 16;
constexpr int DHc = 64;
constexpr int DMc = 1024;
constexpr int BTc = Bc * Tc;   // 4096
constexpr int BHc = Bc * Hc;   // 32

#if __has_builtin(__builtin_amdgcn_tensor_load_to_lds)
#define USE_TDM 1
#else
#define USE_TDM 0
#endif

// Load one WMMA A/B operand (16 bf16 per lane) from a row-major matrix.
// idx already includes row*stride + kb, kb = (lane&16) ? 8 : 0.
#define LDOP16(dst, arrptr, idx) do {                                          \
    v8bf _lo = *(const v8bf*)((arrptr) + (idx));                               \
    v8bf _hi = *(const v8bf*)((arrptr) + (idx) + 16);                          \
    (dst) = __builtin_shufflevector(_lo, _hi,                                  \
            0,1,2,3,4,5,6,7,8,9,10,11,12,13,14,15);                            \
} while (0)

__device__ __forceinline__ v8f wmma_bf16f32(v16bf a, v16bf b, v8f c) {
    return __builtin_amdgcn_wmma_f32_16x16x32_bf16(false, a, false, b,
                                                   (short)0, c, false, false);
}

__device__ __forceinline__ v8f vbroad(float x) {
    v8f r;
#pragma unroll
    for (int i = 0; i < 8; ++i) r[i] = x;
    return r;
}

__device__ __forceinline__ v16bf cvt_a_f32(const float* p) {
    v16bf r;
#pragma unroll
    for (int i = 0; i < 8; ++i) { r[i] = (bf16)p[i]; r[8 + i] = (bf16)p[16 + i]; }
    return r;
}

#if USE_TDM
// Low 32 bits of a generic LDS pointer are the LDS byte offset
// (the aperture lives entirely in ADDR[63:32] per the flat addressing rules).
__device__ __forceinline__ unsigned lds_byte_addr(const void* p) {
    return (unsigned)(unsigned long long)p;
}

// Issue a TDM load of a 2D bf16 tile (tile_d1 rows of tile_d0 contiguous
// elements, rows stride0 elements apart in memory) into LDS at lds_byte,
// with hardware LDS padding of (pad_amount+1) DWORDs every 2^(pad_interval+1)
// DWORDs (i.e. after each tile row).
__device__ __forceinline__ void tdm_load_tile(
    unsigned lds_byte, const void* gptr,
    unsigned tile_d0, unsigned tile_d1, unsigned stride0,
    unsigned pad_interval, unsigned pad_amount)
{
    const unsigned long long ga = (unsigned long long)gptr;
    v4u g0;
    g0[0] = 1u;                                         // count=1, user mode
    g0[1] = lds_byte;                                   // lds_addr
    g0[2] = (unsigned)(ga & 0xffffffffu);               // global_addr[31:0]
    g0[3] = (unsigned)((ga >> 32) & 0x01ffffffu)        // global_addr[56:32]
          | (2u << 30);                                 // type = 2 ("image")
    v8i g1;
    g1[0] = (int)((1u << 16)                            // data_size = 2 bytes
                | (1u << 20)                            // pad_enable
                | (pad_interval << 22)
                | (pad_amount << 25));
    g1[1] = (int)((tile_d0 & 0xffffu) << 16);           // tensor_dim0 lo (== tile_d0)
    g1[2] = (int)(((tile_d0 >> 16) & 0xffffu)           // tensor_dim0 hi
                | ((tile_d1 & 0xffffu) << 16));         // tensor_dim1 lo (== tile_d1)
    g1[3] = (int)(((tile_d1 >> 16) & 0xffffu)
                | ((tile_d0 & 0xffffu) << 16));         // tile_dim0
    g1[4] = (int)(tile_d1 & 0xffffu);                   // tile_dim1 (tile_dim2 = 0)
    g1[5] = (int)stride0;                               // tensor_dim0_stride lo
    g1[6] = 0;
    g1[7] = 0;
    v4i z4 = {0, 0, 0, 0};
#if __has_include(<hip/amd_detail/amd_gfx1250_TDM.h>)
    v8i z8 = {0, 0, 0, 0, 0, 0, 0, 0};                  // amdgpu-toolchain: 6-arg form
    __builtin_amdgcn_tensor_load_to_lds(g0, g1, z4, z4, z8, 0);
#else
    __builtin_amdgcn_tensor_load_to_lds(g0, g1, z4, z4, 0);  // ROCm 7.2: 5-arg form
#endif
}
#endif  // USE_TDM

// ---------------------------------------------------------------------------
// Kernel 1: per-head QKV projection.  q = xh @ Wq^T + bq (etc).
// Grid (BT/128, H), block 256.  Each wave: 16 rows x 64 cols for Q, K, Vt.
// ---------------------------------------------------------------------------
__global__ __launch_bounds__(256) void k_qkv(
    const float* __restrict__ x,
    const float* __restrict__ Wq, const float* __restrict__ bq,
    const float* __restrict__ Wk, const float* __restrict__ bk,
    const float* __restrict__ Wv, const float* __restrict__ bv,
    bf16* __restrict__ Qws, bf16* __restrict__ Kws, bf16* __restrict__ Vtws)
{
    constexpr int WS = 72;  // padded LDS row stride (bf16); 144B: 16B-aligned, bank-clean
    __shared__ bf16 lw[3 * 64 * WS];

    const int tid = threadIdx.x;
    {   // stage Wq/Wk/Wv f32 -> bf16 in LDS
        const int e  = tid >> 2;
        const int d0 = (tid & 3) * 16;
#pragma unroll
        for (int m = 0; m < 3; ++m) {
            const float* W = (m == 0) ? Wq : ((m == 1) ? Wk : Wv);
#pragma unroll
            for (int j = 0; j < 16; ++j)
                lw[m * 64 * WS + e * WS + d0 + j] = (bf16)W[e * 64 + d0 + j];
        }
    }
    __syncthreads();

    const int lane = tid & 31;
    const int wv   = tid >> 5;
    const int h    = blockIdx.y;
    const int row0 = blockIdx.x * 128 + wv * 16;   // [0, 4096)
    const int r    = lane & 15;
    const int kb   = (lane & 16) ? 8 : 0;
    const int bidx = row0 >> 11;
    const int t0   = row0 & (Tc - 1);
    const int bh   = bidx * Hc + h;

    const float* xrow = x + (size_t)(row0 + r) * DMc + h * DHc;
    const v16bf a0 = cvt_a_f32(xrow + kb);
    const v16bf a1 = cvt_a_f32(xrow + 32 + kb);

#pragma unroll
    for (int m = 0; m < 3; ++m) {
        const float* bias = (m == 0) ? bq : ((m == 1) ? bk : bv);
#pragma unroll
        for (int nt = 0; nt < 4; ++nt) {
            const int e = nt * 16 + r;
            v8f acc = vbroad(bias[e]);
            v16bf b0, b1;
            LDOP16(b0, lw, m * 64 * WS + e * WS + kb);
            LDOP16(b1, lw, m * 64 * WS + e * WS + 32 + kb);
            acc = wmma_bf16f32(a0, b0, acc);
            acc = wmma_bf16f32(a1, b1, acc);
            if (m < 2) {
                bf16* dst = (m == 0) ? Qws : Kws;
#pragma unroll
                for (int i = 0; i < 8; ++i) {
                    const int mr = (lane & 16) ? (8 + i) : i;
                    dst[((size_t)bh * Tc + t0 + mr) * DHc + e] = (bf16)acc[i];
                }
            } else {   // store V transposed: Vt[bh][e][t]
#pragma unroll
                for (int i = 0; i < 8; ++i) {
                    const int mr = (lane & 16) ? (8 + i) : i;
                    Vtws[((size_t)bh * DHc + e) * Tc + t0 + mr] = (bf16)acc[i];
                }
            }
        }
    }
}

// ---------------------------------------------------------------------------
// Kernel 2: flash attention.  Grid (T/128, B*H), block 256 (8 waves x 16 q-rows).
// S^T = K·Q^T so the exp'd tiles are already in B-operand layout for
// O^T = V^T·P^T.  K/Vt tiles double-buffered in LDS via the TDM.
// ---------------------------------------------------------------------------
__global__ __launch_bounds__(256) void k_attn(
    const bf16* __restrict__ Qws, const bf16* __restrict__ Kws,
    const bf16* __restrict__ Vtws, bf16* __restrict__ Attn)
{
    constexpr int KS = 88;   // K tile  [32 keys][64 d]  padded stride (bf16)
    constexpr int VS = 56;   // Vt tile [64 d][32 keys]  padded stride (bf16)
    __shared__ bf16 lk[2][32 * KS];
    __shared__ bf16 lv[2][64 * VS];

    const int tid  = threadIdx.x;
    const int lane = tid & 31;
    const int wv   = tid >> 5;
    const int bh   = blockIdx.y;
    const int q0   = blockIdx.x * 128 + wv * 16;
    const int r    = lane & 15;
    const int kb   = (lane & 16) ? 8 : 0;

    const bf16* Qb = Qws  + ((size_t)bh * Tc + q0) * DHc;
    const bf16* Kb = Kws  + (size_t)bh * Tc * DHc;
    const bf16* Vb = Vtws + (size_t)bh * DHc * Tc;

    v16bf qb0, qb1;
    LDOP16(qb0, Qb, r * DHc + kb);
    LDOP16(qb1, Qb, r * DHc + 32 + kb);

    v8f o0 = vbroad(0.f), o1 = vbroad(0.f), o2 = vbroad(0.f), o3 = vbroad(0.f);
    float mrun = -1e30f, lrun = 0.f;
    const float scale = 0.03125f;                   // 1/sqrt(d_model=1024)

    const int nIter = Tc / 32;

#if USE_TDM
    // One wave drives the TDM; workgroup barriers publish the tiles.
    // K row = 64 bf16 = 32 DWORDs  -> pad_interval 4 (2^5=32), pad 12 DW -> stride 88
    // Vt row = 32 bf16 = 16 DWORDs -> pad_interval 3 (2^4=16), pad 12 DW -> stride 56
    if (wv == 0) {
        tdm_load_tile(lds_byte_addr(&lk[0][0]), Kb, 64, 32, 64, 4u, 11u);
        tdm_load_tile(lds_byte_addr(&lv[0][0]), Vb, 32, 64, Tc, 3u, 11u);
    }
#else
    const int skey = tid >> 3, sd = (tid & 7) * 8;
    const int svd  = tid >> 2, sk = (tid & 3) * 8;
    *(v8bf*)&lk[0][skey * KS + sd] = *(const v8bf*)(Kb + (size_t)skey * DHc + sd);
    *(v8bf*)&lv[0][svd * VS + sk]  = *(const v8bf*)(Vb + (size_t)svd * Tc + sk);
#endif

    for (int it = 0; it < nIter; ++it) {
        const int cur = it & 1;
        const int kc  = it * 32;
#if USE_TDM
        if (wv == 0) {
            if (it + 1 < nIter) {
                const int nb = (it + 1) & 1;
                const int nc = kc + 32;
                tdm_load_tile(lds_byte_addr(&lk[nb][0]),
                              Kb + (size_t)nc * DHc, 64, 32, 64, 4u, 11u);
                tdm_load_tile(lds_byte_addr(&lv[nb][0]),
                              Vb + nc, 32, 64, Tc, 3u, 11u);
                __builtin_amdgcn_s_wait_tensorcnt(2);   // current pair complete
            } else {
                __builtin_amdgcn_s_wait_tensorcnt(0);
            }
        }
#else
        if (it + 1 < nIter) {
            const int nb = (it + 1) & 1;
            const int nc = kc + 32;
            *(v8bf*)&lk[nb][skey * KS + sd] = *(const v8bf*)(Kb + (size_t)(nc + skey) * DHc + sd);
            *(v8bf*)&lv[nb][svd * VS + sk]  = *(const v8bf*)(Vb + (size_t)svd * Tc + nc + sk);
        }
#endif
        __syncthreads();

        const bf16* ck = lk[cur];
        const bf16* cv = lv[cur];

        // GEMM1: S^T tiles (keys x q), k-dim = d_head (2 chunks of 32)
        v8f s0 = vbroad(0.f), s1 = vbroad(0.f);
        v16bf ak;
        LDOP16(ak, ck, r * KS + kb);              s0 = wmma_bf16f32(ak, qb0, s0);
        LDOP16(ak, ck, r * KS + 32 + kb);         s0 = wmma_bf16f32(ak, qb1, s0);
        LDOP16(ak, ck, (16 + r) * KS + kb);       s1 = wmma_bf16f32(ak, qb0, s1);
        LDOP16(ak, ck, (16 + r) * KS + 32 + kb);  s1 = wmma_bf16f32(ak, qb1, s1);

        // online softmax: lane owns column q = q0 + r; lane L / L+16 hold
        // complementary key halves -> one xor-16 shuffle finishes a row stat.
        float cmax = -1e30f;
#pragma unroll
        for (int i = 0; i < 8; ++i) {
            s0[i] *= scale; s1[i] *= scale;
            cmax = fmaxf(cmax, fmaxf(s0[i], s1[i]));
        }
        cmax = fmaxf(cmax, __shfl_xor(cmax, 16));
        const float mnew = fmaxf(mrun, cmax);
        const float corr = __expf(mrun - mnew);
        float psum = 0.f;
#pragma unroll
        for (int i = 0; i < 8; ++i) {
            s0[i] = __expf(s0[i] - mnew);
            s1[i] = __expf(s1[i] - mnew);
            psum += s0[i] + s1[i];
        }
        psum += __shfl_xor(psum, 16);
        lrun = lrun * corr + psum;
        mrun = mnew;

        // pack P^T straight into the GEMM2 B-operand layout (no lane movement)
        v16bf pb;
#pragma unroll
        for (int i = 0; i < 8; ++i) { pb[i] = (bf16)s0[i]; pb[8 + i] = (bf16)s1[i]; }

        // rescale O^T accumulators then O^T += V^T x P^T (4 d-tiles of 16)
#pragma unroll
        for (int i = 0; i < 8; ++i) { o0[i]*=corr; o1[i]*=corr; o2[i]*=corr; o3[i]*=corr; }
        v16bf av;
        LDOP16(av, cv, (0 * 16 + r) * VS + kb);  o0 = wmma_bf16f32(av, pb, o0);
        LDOP16(av, cv, (1 * 16 + r) * VS + kb);  o1 = wmma_bf16f32(av, pb, o1);
        LDOP16(av, cv, (2 * 16 + r) * VS + kb);  o2 = wmma_bf16f32(av, pb, o2);
        LDOP16(av, cv, (3 * 16 + r) * VS + kb);  o3 = wmma_bf16f32(av, pb, o3);
        __syncthreads();
    }

    // epilogue: O[q,d] = O^T/l into the [B*T, D_MODEL] activation layout
    const float inv = 1.f / lrun;
    const int   bo  = bh >> 4, ho = bh & 15;
    const size_t arow = ((size_t)bo * Tc + q0 + r) * DMc + ho * DHc;
    const int mo = (lane & 16) ? 8 : 0;
#pragma unroll
    for (int i = 0; i < 8; ++i) {
        Attn[arow + 0 * 16 + mo + i] = (bf16)(o0[i] * inv);
        Attn[arow + 1 * 16 + mo + i] = (bf16)(o1[i] * inv);
        Attn[arow + 2 * 16 + mo + i] = (bf16)(o2[i] * inv);
        Attn[arow + 3 * 16 + mo + i] = (bf16)(o3[i] * inv);
    }
}

// ---------------------------------------------------------------------------
// Kernel 3: output projection  out = Attn @ Wp^T + bp  (f32 output).
// Grid (BT/128, DM/64), block 256.  Wp tile staged bf16 in LDS per k-chunk.
// ---------------------------------------------------------------------------
__global__ __launch_bounds__(256) void k_proj(
    const bf16* __restrict__ Attn, const float* __restrict__ Wp,
    const float* __restrict__ bp, float* __restrict__ out)
{
    constexpr int PS = 56;   // Wp tile [64 n][32 c] padded stride
    __shared__ bf16 lwp[64 * PS];

    const int tid  = threadIdx.x;
    const int lane = tid & 31;
    const int wv   = tid >> 5;
    const int row0 = blockIdx.x * 128 + wv * 16;
    const int n0   = blockIdx.y * 64;
    const int r    = lane & 15;
    const int kb   = (lane & 16) ? 8 : 0;

    v8f acc0 = vbroad(bp[n0 + 0 * 16 + r]);
    v8f acc1 = vbroad(bp[n0 + 1 * 16 + r]);
    v8f acc2 = vbroad(bp[n0 + 2 * 16 + r]);
    v8f acc3 = vbroad(bp[n0 + 3 * 16 + r]);

    const int sn = tid >> 2, sc = (tid & 3) * 8;
    const bf16* arow = Attn + (size_t)(row0 + r) * DMc;

    for (int c0 = 0; c0 < DMc; c0 += 32) {
        {   // stage Wp[n0..n0+63][c0..c0+31] f32 -> bf16
            const float* wp = Wp + (size_t)(n0 + sn) * DMc + c0 + sc;
            v8bf wb;
#pragma unroll
            for (int j = 0; j < 8; ++j) wb[j] = (bf16)wp[j];
            *(v8bf*)&lwp[sn * PS + sc] = wb;
        }
        __syncthreads();
        if (c0 + 32 < DMc)   // gfx1250 global_prefetch_b8
            __builtin_prefetch(Wp + (size_t)(n0 + sn) * DMc + c0 + 32 + sc, 0, 1);

        v16bf a;
        LDOP16(a, arow, c0 + kb);
        v16bf wb0;
        LDOP16(wb0, lwp, (0 * 16 + r) * PS + kb);  acc0 = wmma_bf16f32(a, wb0, acc0);
        LDOP16(wb0, lwp, (1 * 16 + r) * PS + kb);  acc1 = wmma_bf16f32(a, wb0, acc1);
        LDOP16(wb0, lwp, (2 * 16 + r) * PS + kb);  acc2 = wmma_bf16f32(a, wb0, acc2);
        LDOP16(wb0, lwp, (3 * 16 + r) * PS + kb);  acc3 = wmma_bf16f32(a, wb0, acc3);
        __syncthreads();
    }

    const int mo = (lane & 16) ? 8 : 0;
#pragma unroll
    for (int i = 0; i < 8; ++i) {
        const size_t orow = (size_t)(row0 + mo + i) * DMc + n0;
        out[orow + 0 * 16 + r] = acc0[i];
        out[orow + 1 * 16 + r] = acc1[i];
        out[orow + 2 * 16 + r] = acc2[i];
        out[orow + 3 * 16 + r] = acc3[i];
    }
}

// ---------------------------------------------------------------------------

extern "C" void kernel_launch(void* const* d_in, const int* in_sizes, int n_in,
                              void* d_out, int out_size, void* d_ws, size_t ws_size,
                              hipStream_t stream) {
    (void)in_sizes; (void)n_in; (void)out_size; (void)ws_size;
    const float* x  = (const float*)d_in[0];
    const float* Wq = (const float*)d_in[1];
    const float* bq = (const float*)d_in[2];
    const float* Wk = (const float*)d_in[3];
    const float* bk = (const float*)d_in[4];
    const float* Wv = (const float*)d_in[5];
    const float* bv = (const float*)d_in[6];
    const float* Wp = (const float*)d_in[7];
    const float* bp = (const float*)d_in[8];
    float* out = (float*)d_out;

    const size_t NQ = (size_t)BHc * Tc * DHc;   // 4 Mi elements per tensor
    bf16* Qws  = (bf16*)d_ws;
    bf16* Kws  = Qws + NQ;
    bf16* Vtws = Kws + NQ;
    bf16* Attn = Vtws + NQ;                     // [4096][1024] bf16

    k_qkv <<<dim3(BTc / 128, Hc),       256, 0, stream>>>(x, Wq, bq, Wk, bk, Wv, bv,
                                                          Qws, Kws, Vtws);
    k_attn<<<dim3(Tc / 128, BHc),       256, 0, stream>>>(Qws, Kws, Vtws, Attn);
    k_proj<<<dim3(BTc / 128, DMc / 64), 256, 0, stream>>>(Attn, Wp, bp, out);
}
